// SpatialOnlyGNN_19911468384347
// MI455X (gfx1250) — matrix operand order repaired and checked by reading, not compiled, not run on hardware
//
#include <hip/hip_runtime.h>
#include <hip/hip_bf16.h>

typedef __attribute__((ext_vector_type(16))) _Float16 v16h;
typedef __attribute__((ext_vector_type(8)))  _Float16 v8h;
typedef __attribute__((ext_vector_type(8)))  float    v8f;

#define Bc   4
#define Nc   20000
#define Tc   24
#define Cc   2
#define Ec   320000
#define Hc   4
#define Fc   16
#define HIDc 64
#define TFc  14
#define BNc  (Bc * Nc)          // 80000
#define BEc  (Bc * Ec)          // 1280000
#define LN_EPS 1e-5f
#define NEG_SLOPE 0.2f

// ---------------------------------------------------------------- utilities
__device__ __forceinline__ void atomicMaxFloat(float* addr, float val) {
    // signed-int max for positives, unsigned min of bit pattern for negatives
    if (val >= 0.f)
        atomicMax((int*)addr, __float_as_int(val));
    else
        atomicMin((unsigned int*)addr, (unsigned int)__float_as_int(val));
}

// ---------------------------------------------------------------- 1) temporal mean
__global__ void k_mean(const float* __restrict__ x_seq, float* __restrict__ h) {
    int n = blockIdx.x * blockDim.x + threadIdx.x;
    if (n >= BNc) return;
    const float* xp = x_seq + (size_t)n * (Tc * Cc);
    float s0 = 0.f, s1 = 0.f;
    #pragma unroll
    for (int t = 0; t < Tc; ++t) { s0 += xp[t * 2]; s1 += xp[t * 2 + 1]; }
    h[n * 2 + 0] = s0 * (1.f / Tc);
    h[n * 2 + 1] = s1 * (1.f / Tc);
}

// ---------------------------------------------------------------- 2) X0 = h@W0, SKIP = h@skip_W + skip_b  (C=2 analytic)
__global__ void k_x0skip(const float* __restrict__ h,
                         const float* __restrict__ W0,
                         const float* __restrict__ skW, const float* __restrict__ skb,
                         float* __restrict__ X, float* __restrict__ SKIP) {
    int id = blockIdx.x * blockDim.x + threadIdx.x;
    if (id >= BNc * HIDc) return;
    int n = id >> 6, j = id & 63;
    float h0 = h[n * 2], h1 = h[n * 2 + 1];
    X[id]    = h0 * W0[j]  + h1 * W0[HIDc + j];
    SKIP[id] = h0 * skW[j] + h1 * skW[HIDc + j] + skb[j];
}

// ---------------------------------------------------------------- 3) edge coefficients ce[h] = sum_f We[h*F+f]*att_e[h*F+f]
__global__ void k_ce(const float* __restrict__ We0, const float* __restrict__ ae0,
                     const float* __restrict__ We1, const float* __restrict__ ae1,
                     float* __restrict__ CE) {
    int t = threadIdx.x;
    if (t >= 2 * Hc) return;
    const float* We = (t < Hc) ? We0 : We1;
    const float* ae = (t < Hc) ? ae0 : ae1;
    int hh = t & 3;
    float s = 0.f;
    #pragma unroll
    for (int f = 0; f < Fc; ++f) s += We[hh * Fc + f] * ae[hh * Fc + f];
    CE[t] = s;
}

// ---------------------------------------------------------------- 4) weight conversion: f32 -> f16 packed into WMMA B-fragment order
// W1P layout:  [ntile(4)][kstep(2)][lane(32)][16 halfs]  (lane's v16h contiguous)
// HWP layout:  [kstep(2)][lane(32)][16 halfs]            (head_W zero-padded to 16 cols)
// B-fragment order used here (contiguous k, same as load_afrag): lane l -> col = l&15, K = kstep*32 + (l>>4)*16 + i
__global__ void k_cvt(const float* __restrict__ W1, const float* __restrict__ headW,
                      _Float16* __restrict__ W1P, _Float16* __restrict__ HWP) {
    int id = blockIdx.x * blockDim.x + threadIdx.x;
    if (id < HIDc * HIDc) {
        int i     = id & 15;
        int lane  = (id >> 4) & 31;
        int kstep = (id >> 9) & 1;
        int ntile = id >> 10;                       // 0..3
        int col   = ntile * 16 + (lane & 15);
        int krow  = kstep * 32 + (lane >> 4) * 16 + i;
        W1P[id] = (_Float16)W1[krow * HIDc + col];
    } else if (id < HIDc * HIDc + 2 * 32 * 16) {
        int j     = id - HIDc * HIDc;
        int i     = j & 15;
        int lane  = (j >> 4) & 31;
        int kstep = j >> 9;                         // 0..1
        int col   = lane & 15;
        int krow  = kstep * 32 + (lane >> 4) * 16 + i;
        HWP[j] = (col < TFc) ? (_Float16)headW[krow * TFc + col] : (_Float16)0.f;
    }
}

// ---------------------------------------------------------------- 5) per-node attention prep (L2-resident)
__global__ void k_attnprep(const float* __restrict__ X,
                           const float* __restrict__ atts, const float* __restrict__ attd,
                           float* __restrict__ AS, float* __restrict__ AD,
                           float* __restrict__ MX, float* __restrict__ DEN,
                           float* __restrict__ OUT) {
    int t = blockIdx.x * blockDim.x + threadIdx.x;
    if (t >= BNc * Hc) return;
    int n = t >> 2, hh = t & 3;
    const float* xr = X + (size_t)n * HIDc + hh * Fc;
    float s = 0.f, d = 0.f;
    #pragma unroll
    for (int f = 0; f < Fc; ++f) {
        float x = xr[f];
        s += x * atts[hh * Fc + f];
        d += x * attd[hh * Fc + f];
    }
    AS[t] = s; AD[t] = d;
    MX[t] = -__builtin_inff();
    DEN[t] = 0.f;
    float* o = OUT + (size_t)n * HIDc + hh * Fc;
    #pragma unroll
    for (int f = 0; f < Fc; ++f) o[f] = 0.f;
}

// ---------------------------------------------------------------- 6) edge pass 1: alpha + segment max
__global__ void k_edge_max(const int* __restrict__ ei, const float* __restrict__ ew,
                           const float* __restrict__ AS, const float* __restrict__ AD,
                           const float* __restrict__ CE,
                           float* __restrict__ MX, float* __restrict__ ALPHA) {
    int t = blockIdx.x * blockDim.x + threadIdx.x;
    if (t >= BEc * Hc) return;
    int hh = t & 3, e = t >> 2;
    int b = e / Ec, eo = e % Ec;
    int s = ei[eo] + b * Nc;
    int d = ei[Ec + eo] + b * Nc;
    float a = AS[s * Hc + hh] + AD[d * Hc + hh] + ew[eo] * CE[hh];
    a = (a > 0.f) ? a : NEG_SLOPE * a;
    ALPHA[t] = a;
    atomicMaxFloat(&MX[d * Hc + hh], a);
}

// ---------------------------------------------------------------- 7) edge pass 2: exp + segment sum
__global__ void k_edge_exp(const int* __restrict__ ei,
                           const float* __restrict__ MX, float* __restrict__ DEN,
                           float* __restrict__ ALPHA) {
    int t = blockIdx.x * blockDim.x + threadIdx.x;
    if (t >= BEc * Hc) return;
    int hh = t & 3, e = t >> 2;
    int b = e / Ec, eo = e % Ec;
    int d = ei[Ec + eo] + b * Nc;
    float ex = __expf(ALPHA[t] - MX[d * Hc + hh]);
    ALPHA[t] = ex;
    atomicAdd(&DEN[d * Hc + hh], ex);
}

// ---------------------------------------------------------------- 8) edge pass 3: weighted scatter (atomics stay in L2)
__global__ void k_edge_scatter(const int* __restrict__ ei,
                               const float* __restrict__ ALPHA, const float* __restrict__ DEN,
                               const float* __restrict__ X, float* __restrict__ OUT) {
    int t = blockIdx.x * blockDim.x + threadIdx.x;
    if (t >= BEc * Hc) return;
    int hh = t & 3, e = t >> 2;
    int b = e / Ec, eo = e % Ec;
    int s = ei[eo] + b * Nc;
    int d = ei[Ec + eo] + b * Nc;
    const float4* xr = (const float4*)(X + (size_t)s * HIDc + hh * Fc);
    __builtin_prefetch(xr, 0, 1);                    // global_prefetch_b8
    float w = ALPHA[t] / DEN[d * Hc + hh];
    float* o = OUT + (size_t)d * HIDc + hh * Fc;
    #pragma unroll
    for (int q = 0; q < 4; ++q) {
        float4 x4 = xr[q];
        atomicAdd(o + 4 * q + 0, x4.x * w);
        atomicAdd(o + 4 * q + 1, x4.y * w);
        atomicAdd(o + 4 * q + 2, x4.z * w);
        atomicAdd(o + 4 * q + 3, x4.w * w);
    }
}

// ---------------------------------------------------------------- 9) LayerNorm + residual + ReLU; emit f32 + f16 copies
__global__ void k_post(const float* __restrict__ OUT, const float* __restrict__ bias,
                       const float* __restrict__ g, const float* __restrict__ bln,
                       const float* __restrict__ resid,
                       float* __restrict__ Hout, _Float16* __restrict__ Hf) {
    __shared__ float sm[HIDc];
    int n = blockIdx.x, j = threadIdx.x;
    float val = OUT[(size_t)n * HIDc + j] + bias[j];
    sm[j] = val; __syncthreads();
    #pragma unroll
    for (int s = 32; s > 0; s >>= 1) { if (j < s) sm[j] += sm[j + s]; __syncthreads(); }
    float mean = sm[0] * (1.f / HIDc); __syncthreads();
    float dv = val - mean;
    sm[j] = dv * dv; __syncthreads();
    #pragma unroll
    for (int s = 32; s > 0; s >>= 1) { if (j < s) sm[j] += sm[j + s]; __syncthreads(); }
    float var = sm[0] * (1.f / HIDc);
    float y = dv * __frsqrt_rn(var + LN_EPS) * g[j] + bln[j] + resid[(size_t)n * HIDc + j];
    y = fmaxf(y, 0.f);
    Hout[(size_t)n * HIDc + j] = y;
    Hf[(size_t)n * HIDc + j] = (_Float16)y;
}

// ---------------------------------------------------------------- A-fragment loader: two contiguous B128 loads + concat
// 16-bit A 16x32 layout (contiguous, matching packed B): lane group 0 holds K={0..15}, group 1 holds K={16..31}
__device__ __forceinline__ v16h load_afrag(const _Float16* __restrict__ Arow,
                                           int ks, int gsel) {
    v8h lo = *(const v8h*)(Arow + ks + (gsel ? 16 : 0));
    v8h hi = *(const v8h*)(Arow + ks + (gsel ? 24 : 8));
    return __builtin_shufflevector(lo, hi, 0, 1, 2, 3, 4, 5, 6, 7,
                                           8, 9, 10, 11, 12, 13, 14, 15);
}

// ---------------------------------------------------------------- 10) WMMA GEMM: X = H1f16 @ W1f16  (80000x64 @ 64x64)
// 4 waves/block, one 16x16 N-tile per wave, K=64 as two 16x16x32 steps.
// B fragments pre-packed: each lane reads one contiguous 32B chunk per k-step.
__global__ void k_gemm64(const _Float16* __restrict__ Ah, const _Float16* __restrict__ Bp,
                         float* __restrict__ Xo) {
    int wave = threadIdx.x >> 5;        // 0..3 -> N tile
    int lane = threadIdx.x & 31;
    int rowBase = blockIdx.x * 16;
    int row = lane & 15, gsel = lane >> 4;
    const _Float16* Arow = Ah + (size_t)(rowBase + row) * HIDc;
    v8f acc = {};
    #pragma unroll
    for (int kstep = 0; kstep < 2; ++kstep) {
        v16h a = load_afrag(Arow, kstep * 32, gsel);
        v16h b = *(const v16h*)(Bp + ((size_t)(wave * 2 + kstep) * 32 + lane) * 16);
        acc = __builtin_amdgcn_wmma_f32_16x16x32_f16(false, a, false, b, (short)0, acc,
                                                     false, false);
    }
    #pragma unroll
    for (int r = 0; r < 8; ++r)        // D: M = r + 8*lanegroup, N = lane&15
        Xo[(size_t)(rowBase + r + 8 * gsel) * HIDc + wave * 16 + row] = acc[r];
}

// ---------------------------------------------------------------- 11) WMMA head: out = H2f16 @ headWf16(64x16pad) + b, cols<14
__global__ void k_head(const _Float16* __restrict__ Ah, const _Float16* __restrict__ Bp,
                       const float* __restrict__ hb, float* __restrict__ out) {
    int lane = threadIdx.x & 31;
    int rowBase = blockIdx.x * 16;
    int row = lane & 15, gsel = lane >> 4;
    const _Float16* Arow = Ah + (size_t)(rowBase + row) * HIDc;
    v8f acc = {};
    #pragma unroll
    for (int kstep = 0; kstep < 2; ++kstep) {
        v16h a = load_afrag(Arow, kstep * 32, gsel);
        v16h b = *(const v16h*)(Bp + ((size_t)kstep * 32 + lane) * 16);
        acc = __builtin_amdgcn_wmma_f32_16x16x32_f16(false, a, false, b, (short)0, acc,
                                                     false, false);
    }
    if (row < TFc) {
        float bb = hb[row];
        #pragma unroll
        for (int r = 0; r < 8; ++r)
            out[(size_t)(rowBase + r + 8 * gsel) * TFc + row] = acc[r] + bb;
    }
}

// ---------------------------------------------------------------- launch
extern "C" void kernel_launch(void* const* d_in, const int* in_sizes, int n_in,
                              void* d_out, int out_size, void* d_ws, size_t ws_size,
                              hipStream_t stream) {
    const float* x_seq  = (const float*)d_in[0];
    const int*   eidx   = (const int*)  d_in[1];
    const float* ew     = (const float*)d_in[2];
    const float* W0     = (const float*)d_in[3];
    const float* We0    = (const float*)d_in[4];
    const float* atts0  = (const float*)d_in[5];
    const float* attd0  = (const float*)d_in[6];
    const float* atte0  = (const float*)d_in[7];
    const float* b0     = (const float*)d_in[8];
    const float* lng0   = (const float*)d_in[9];
    const float* lnb0   = (const float*)d_in[10];
    const float* W1     = (const float*)d_in[11];
    const float* We1    = (const float*)d_in[12];
    const float* atts1  = (const float*)d_in[13];
    const float* attd1  = (const float*)d_in[14];
    const float* atte1  = (const float*)d_in[15];
    const float* b1     = (const float*)d_in[16];
    const float* lng1   = (const float*)d_in[17];
    const float* lnb1   = (const float*)d_in[18];
    const float* skW    = (const float*)d_in[19];
    const float* skb    = (const float*)d_in[20];
    const float* headW  = (const float*)d_in[21];
    const float* headb  = (const float*)d_in[22];
    float* out = (float*)d_out;

    // workspace carve-up (all segments 64B-aligned by construction)
    float* ws    = (float*)d_ws;
    float* h     = ws;                      // BN*2
    float* SKIP  = h     + (size_t)BNc * 2; // BN*64 (reused as H2 storage)
    float* X     = SKIP  + (size_t)BNc * HIDc;
    float* OUT   = X     + (size_t)BNc * HIDc;
    float* H1    = OUT   + (size_t)BNc * HIDc;
    float* AS    = H1    + (size_t)BNc * HIDc;
    float* AD    = AS    + (size_t)BNc * Hc;
    float* MX    = AD    + (size_t)BNc * Hc;
    float* DEN   = MX    + (size_t)BNc * Hc;
    float* ALPHA = DEN   + (size_t)BNc * Hc;   // BE*4
    float* CE    = ALPHA + (size_t)BEc * Hc;   // 8 (+pad)
    _Float16* W1P = (_Float16*)(CE + 16);      // 4096 (fragment-packed W1)
    _Float16* HWP = W1P + HIDc * HIDc;         // 1024 (fragment-packed padded head_W)
    _Float16* HF  = HWP + 2 * 32 * 16;         // BN*64 halfs

    const int TB = 256;
    int gNode  = (BNc + TB - 1) / TB;
    int gNJ    = (BNc * HIDc + TB - 1) / TB;
    int gNH    = (BNc * Hc + TB - 1) / TB;
    int gEH    = (BEc * Hc + TB - 1) / TB;
    int gTile  = BNc / 16;                     // 5000

    k_mean<<<gNode, TB, 0, stream>>>(x_seq, h);
    k_x0skip<<<gNJ, TB, 0, stream>>>(h, W0, skW, skb, X, SKIP);
    k_ce<<<1, 32, 0, stream>>>(We0, atte0, We1, atte1, CE);
    k_cvt<<<(HIDc * HIDc + 2 * 32 * 16 + TB - 1) / TB, TB, 0, stream>>>(W1, headW, W1P, HWP);

    // ---- layer 0
    k_attnprep<<<gNH, TB, 0, stream>>>(X, atts0, attd0, AS, AD, MX, DEN, OUT);
    k_edge_max<<<gEH, TB, 0, stream>>>(eidx, ew, AS, AD, CE, MX, ALPHA);
    k_edge_exp<<<gEH, TB, 0, stream>>>(eidx, MX, DEN, ALPHA);
    k_edge_scatter<<<gEH, TB, 0, stream>>>(eidx, ALPHA, DEN, X, OUT);
    k_post<<<BNc, HIDc, 0, stream>>>(OUT, b0, lng0, lnb0, SKIP, H1, HF);

    // ---- layer 1 (WMMA GEMM for the 64x64 linear)
    k_gemm64<<<gTile, 128, 0, stream>>>(HF, W1P, X);
    k_attnprep<<<gNH, TB, 0, stream>>>(X, atts1, attd1, AS, AD, MX, DEN, OUT);
    k_edge_max<<<gEH, TB, 0, stream>>>(eidx, ew, AS, AD, CE + Hc, MX, ALPHA);
    k_edge_exp<<<gEH, TB, 0, stream>>>(eidx, MX, DEN, ALPHA);
    k_edge_scatter<<<gEH, TB, 0, stream>>>(eidx, ALPHA, DEN, X, OUT);
    k_post<<<BNc, HIDc, 0, stream>>>(OUT, b1, lng1, lnb1, H1, /*H2=*/SKIP, HF);

    // ---- forecasting head (WMMA)
    k_head<<<gTile, 32, 0, stream>>>(HF, HWP, headb, out);
}
